// GCN_Sparse_Value_30528627540629
// MI455X (gfx1250) — compile-verified, matrix operand order checked by reading, not run on hardware
//
#include <hip/hip_runtime.h>
#include <hip/hip_bf16.h>

#define NODES_F 128              // feature width (NIN == NH == 128)
#define LDSTRIDE 136             // padded K-stride for LDS W image (bank-conflict avoidance)
#define EDGES_PER_WAVE 8

typedef __attribute__((ext_vector_type(16))) __bf16 v16bf;
typedef __attribute__((ext_vector_type(8)))  __bf16 v8bf;
typedef __attribute__((ext_vector_type(8)))  float  v8f;
typedef __attribute__((ext_vector_type(4)))  float  f32x4;

// Split f32 into bf16 hi + bf16 lo using native conversions (gfx1250 has bf16 cvt).
__device__ __forceinline__ void split_bf16(float f, __bf16& hi, __bf16& lo) {
    hi = (__bf16)f;
    lo = (__bf16)(f - (float)hi);
}

// Y[M x 128] = X[M x 128] @ W[128 x 128], f32 in/out, bf16x3 WMMA internally.
// Block: 256 threads (8 waves). Wave w computes rows [blk*128 + w*16, +16) x all 128 cols.
__global__ __launch_bounds__(256)
void gemm_bf16x3(const float* __restrict__ X, const float* __restrict__ W,
                 float* __restrict__ Y, int M) {
    __shared__ __bf16 whi[NODES_F * LDSTRIDE];
    __shared__ __bf16 wlo[NODES_F * LDSTRIDE];

    const int tid = threadIdx.x;

    // Cooperative load of W[k][n] -> transposed bf16 hi/lo LDS image at [n*LDSTRIDE + k].
    for (int i = tid; i < NODES_F * NODES_F; i += 256) {
        int k = i >> 7;
        int n = i & 127;
        __bf16 h, l;
        split_bf16(W[i], h, l);
        whi[n * LDSTRIDE + k] = h;
        wlo[n * LDSTRIDE + k] = l;
    }
    __syncthreads();

    const int wave     = tid >> 5;
    const int lane     = tid & 31;
    const int laneHalf = (lane >> 4) & 1;     // 0: lanes 0-15, 1: lanes 16-31
    const int lmod     = lane & 15;
    const int row0     = blockIdx.x * 128 + wave * 16;
    if (row0 >= M) return;                    // M % 16 == 0, so per-wave tiles are all-or-nothing

    v8f acc[8] = {};                          // 8 N-tiles of 16x16 f32

    const float* arow = X + (size_t)(row0 + lmod) * NODES_F;

    for (int kb = 0; kb < NODES_F; kb += 32) {
        // ---- A fragment (16x32 bf16 layout, ISA 7.12.2): lane holds K runs
        // [k0 .. k0+7] in elements 0..7 and [k0+16 .. k0+23] in elements 8..15,
        // where k0 = kb + 8*laneHalf.
        const int k0 = kb + laneHalf * 8;
        f32x4 a0 = *(const f32x4*)(arow + k0);
        f32x4 a1 = *(const f32x4*)(arow + k0 + 4);
        f32x4 a2 = *(const f32x4*)(arow + k0 + 16);
        f32x4 a3 = *(const f32x4*)(arow + k0 + 20);
        float af[16];
        #pragma unroll
        for (int e = 0; e < 4; ++e) {
            af[e]      = a0[e];
            af[e + 4]  = a1[e];
            af[e + 8]  = a2[e];
            af[e + 12] = a3[e];
        }
        v16bf ahi, alo;
        #pragma unroll
        for (int e = 0; e < 16; ++e) {
            __bf16 h, l;
            split_bf16(af[e], h, l);
            ahi[e] = h;
            alo[e] = l;
        }

        // ---- 8 N-tiles: B fragment from transposed LDS (column n contiguous in K)
        #pragma unroll
        for (int t = 0; t < 8; ++t) {
            const int n = t * 16 + lmod;
            const __bf16* bh = &whi[n * LDSTRIDE + k0];
            const __bf16* bl = &wlo[n * LDSTRIDE + k0];
            v8bf bh0 = *(const v8bf*)(bh);
            v8bf bh1 = *(const v8bf*)(bh + 16);
            v8bf bl0 = *(const v8bf*)(bl);
            v8bf bl1 = *(const v8bf*)(bl + 16);
            v16bf bhi = __builtin_shufflevector(bh0, bh1, 0,1,2,3,4,5,6,7,8,9,10,11,12,13,14,15);
            v16bf blo = __builtin_shufflevector(bl0, bl1, 0,1,2,3,4,5,6,7,8,9,10,11,12,13,14,15);

            acc[t] = __builtin_amdgcn_wmma_f32_16x16x32_bf16(false, ahi, false, bhi,
                                                             (short)0, acc[t], false, false);
            acc[t] = __builtin_amdgcn_wmma_f32_16x16x32_bf16(false, ahi, false, blo,
                                                             (short)0, acc[t], false, false);
            acc[t] = __builtin_amdgcn_wmma_f32_16x16x32_bf16(false, alo, false, bhi,
                                                             (short)0, acc[t], false, false);
        }
    }

    // ---- Store: C/D layout (ISA 7.12.2): VGPR v, lane l -> row v + 8*laneHalf, col l&15.
    #pragma unroll
    for (int t = 0; t < 8; ++t) {
        #pragma unroll
        for (int v = 0; v < 8; ++v) {
            Y[(size_t)(row0 + v + laneHalf * 8) * NODES_F + t * 16 + lmod] = acc[t][v];
        }
    }
}

// COO SpMM scatter exploiting sorted edge_row: each wave walks EDGES_PER_WAVE
// consecutive edges, accumulating lane-local partial sums while the row id is
// unchanged (wave-uniform branch), flushing atomics only on row transitions.
__global__ __launch_bounds__(256)
void spmm_scatter(const int* __restrict__ erow, const int* __restrict__ ecol,
                  const float* __restrict__ eval, const float* __restrict__ X,
                  float* __restrict__ Y, int nE) {
    int wid  = (blockIdx.x * 256 + threadIdx.x) >> 5;
    int lane = threadIdx.x & 31;
    int e0   = wid * EDGES_PER_WAVE;
    if (e0 >= nE) return;
    int eEnd = min(e0 + EDGES_PER_WAVE, nE);

    int   cur = erow[e0];
    f32x4 acc = (f32x4){0.f, 0.f, 0.f, 0.f};

    for (int e = e0; e < eEnd; ++e) {
        int r = erow[e];                 // wave-uniform
        if (r != cur) {
            float* yp = Y + (size_t)cur * NODES_F + lane * 4;
            atomicAdd(yp + 0, acc[0]);
            atomicAdd(yp + 1, acc[1]);
            atomicAdd(yp + 2, acc[2]);
            atomicAdd(yp + 3, acc[3]);
            acc = (f32x4){0.f, 0.f, 0.f, 0.f};
            cur = r;
        }
        float v = eval[e];
        int   c = ecol[e];
        f32x4 x = *(const f32x4*)(X + (size_t)c * NODES_F + lane * 4);
        acc[0] += v * x[0];
        acc[1] += v * x[1];
        acc[2] += v * x[2];
        acc[3] += v * x[3];
    }
    float* yp = Y + (size_t)cur * NODES_F + lane * 4;
    atomicAdd(yp + 0, acc[0]);
    atomicAdd(yp + 1, acc[1]);
    atomicAdd(yp + 2, acc[2]);
    atomicAdd(yp + 3, acc[3]);
}

__global__ __launch_bounds__(256)
void zero_f32(float* __restrict__ p, int n4) {     // n4 = count of float4
    int i = blockIdx.x * 256 + threadIdx.x;
    if (i < n4) ((f32x4*)p)[i] = (f32x4){0.f, 0.f, 0.f, 0.f};
}

__global__ __launch_bounds__(256)
void bias_relu4(float* __restrict__ h, const float* __restrict__ b, int n4) {
    int i = blockIdx.x * 256 + threadIdx.x;
    if (i < n4) {
        f32x4 x  = ((f32x4*)h)[i];
        f32x4 bb = *(const f32x4*)(b + ((i * 4) & 127));
        #pragma unroll
        for (int j = 0; j < 4; ++j) {
            float v = x[j] + bb[j];
            x[j] = v > 0.f ? v : 0.f;
        }
        ((f32x4*)h)[i] = x;
    }
}

__global__ __launch_bounds__(256)
void bias_init4(float* __restrict__ y, const float* __restrict__ b, int n4) {
    int i = blockIdx.x * 256 + threadIdx.x;
    if (i < n4) ((f32x4*)y)[i] = *(const f32x4*)(b + ((i * 4) & 127));
}

extern "C" void kernel_launch(void* const* d_in, const int* in_sizes, int n_in,
                              void* d_out, int out_size, void* d_ws, size_t ws_size,
                              hipStream_t stream) {
    const float* X    = (const float*)d_in[0];
    const int*   erow = (const int*)  d_in[1];
    const int*   ecol = (const int*)  d_in[2];
    const float* ev   = (const float*)d_in[3];
    const float* W1   = (const float*)d_in[4];
    const float* b1   = (const float*)d_in[5];
    const float* W3   = (const float*)d_in[6];
    const float* b3   = (const float*)d_in[7];
    float* out = (float*)d_out;

    const int M  = in_sizes[0] / NODES_F;   // 50000 nodes
    const int nE = in_sizes[1];             // 500000 edges
    const int NF = M * NODES_F;

    float* t = (float*)d_ws;                // GEMM result buffer  [M x 128]
    float* h = t + (size_t)NF;              // hidden buffer       [M x 128]

    dim3 blk(256);
    int gemmGrid = (M + 127) / 128;
    int vecGrid  = (NF / 4 + 255) / 256;
    int nWaves   = (nE + EDGES_PER_WAVE - 1) / EDGES_PER_WAVE;
    int edgeGrid = (nWaves + 7) / 8;        // 8 waves per block

    // h = relu(spmm(adj, X @ W1) + b1)
    gemm_bf16x3 <<<gemmGrid, blk, 0, stream>>>(X, W1, t, M);
    zero_f32    <<<vecGrid,  blk, 0, stream>>>(h, NF / 4);
    spmm_scatter<<<edgeGrid, blk, 0, stream>>>(erow, ecol, ev, t, h, nE);
    bias_relu4  <<<vecGrid,  blk, 0, stream>>>(h, b1, NF / 4);

    // out = spmm(adj, h @ W3) + b3
    gemm_bf16x3 <<<gemmGrid, blk, 0, stream>>>(h, W3, t, M);
    bias_init4  <<<vecGrid,  blk, 0, stream>>>(out, b3, NF / 4);
    spmm_scatter<<<edgeGrid, blk, 0, stream>>>(erow, ecol, ev, t, out, nE);
}
